// VITContrastiveLoss_72713796321725
// MI455X (gfx1250) — matrix operand order repaired and checked by reading, not compiled, not run on hardware
//
#include <hip/hip_runtime.h>
#include <hip/hip_bf16.h>

// ---------------------------------------------------------------------------
// SimCLR / NT-Xent contrastive loss, fused for MI455X (gfx1250, wave32).
//   loss = mean_i [ logsumexp_{j!=i} (e_i . e_j / T) - (e_i . e_{(i+B)%N} / T) ]
// GEMM (34.4 GFLOP) on v_wmma_f32_16x16x32_f16, f32 accumulate.
// B panels staged in LDS by the Tensor Data Mover (tensor_load_to_lds),
// double-buffered: TDM of panel p+1 overlaps WMMA compute on panel p.
// logsumexp uses the analytic bound max = 1/T (unit vectors): no online max.
// The diagonal can only fall in the one 64-aligned panel with cpan == rb, so
// the panel body is instantiated twice (masked / unmasked) as straight-line
// code: the hot path is compare-free and fully pipelineable.
// ---------------------------------------------------------------------------

#define N_ROWS  8192
#define DIM     256
#define HALF_B  4096
#define CSPLIT  8                  // column chunks per row-block (1024 cols each)
#define PANEL   64                 // columns staged per TDM panel (32 KB)
#define SCALE   (1.0f / 0.07f)     // 1/temperature

typedef __attribute__((ext_vector_type(16))) _Float16 v16h;
typedef __attribute__((ext_vector_type(8)))  _Float16 v8h;
typedef __attribute__((ext_vector_type(8)))  float    v8f;
typedef unsigned int u32x4 __attribute__((ext_vector_type(4)));
typedef int          i32x8 __attribute__((ext_vector_type(8)));
typedef int          i32x4 __attribute__((ext_vector_type(4)));

// ---------------- kernel 1: L2-normalize rows, emit f16 copy ----------------
__global__ void vit_normalize_kernel(const float* __restrict__ in,
                                     _Float16* __restrict__ e16) {
  __shared__ float red[256];
  const int row = blockIdx.x;
  const int t   = threadIdx.x;           // t == feature index (DIM == 256)
  float x = in[(size_t)row * DIM + t];
  red[t] = x * x;
  __syncthreads();
  for (int s = 128; s > 0; s >>= 1) {
    if (t < s) red[t] += red[t + s];
    __syncthreads();
  }
  const float inv = 1.0f / fmaxf(sqrtf(red[0]), 1e-12f);   // torch F.normalize eps
  e16[(size_t)row * DIM + t] = (_Float16)(x * inv);
}

// ---- TDM: DMA a PANELx256-half row-major panel of e16 into LDS ------------
// D# group0: count=1 | lds_addr | global_addr | type=2 ("image")
// D# group1: data_size=2B, tensor 256 x 8192, tile 256 x PANEL, stride 256
// 6-arg builtin on this toolchain: (g0, g1, g2, g3, g4, cpol).
__device__ __forceinline__ void tdm_load_panel(const _Float16* gsrc,
                                               unsigned lds_off_bytes) {
  const unsigned long long ga = (unsigned long long)(uintptr_t)gsrc;
  u32x4 g0;
  g0[0] = 1u;                                        // count=1 (valid), user mode
  g0[1] = lds_off_bytes;                             // lds_addr (bytes)
  g0[2] = (unsigned)(ga & 0xffffffffu);              // global_addr[31:0]
  g0[3] = (unsigned)((ga >> 32) & 0x01ffffffu)       // global_addr[56:32]
        | 0x80000000u;                               // type=2 in bits [127:126]
  i32x8 g1;
  g1[0] = (1 << 16);            // wg_mask=0 (no cluster), data_size=1 -> 2 bytes
  g1[1] = (int)(DIM << 16);     // tensor_dim0[15:0]=256 in bits [31:16]
  g1[2] = (int)(N_ROWS << 16);  // tensor_dim0 hi=0 | tensor_dim1[15:0]=8192
  g1[3] = (int)(DIM << 16);     // tensor_dim1 hi=0 | tile_dim0=256
  g1[4] = PANEL;                // tile_dim1=PANEL | tile_dim2=0 (2-D tile)
  g1[5] = DIM;                  // tensor_dim0_stride[31:0]=256
  g1[6] = 0;                    // stride0 hi | tensor_dim1_stride lo (unused, 2-D)
  g1[7] = 0;
  const i32x4 z4 = {0, 0, 0, 0};               // groups 2/3: unused (2-D tile)
  const i32x8 z8 = {0, 0, 0, 0, 0, 0, 0, 0};   // extra group: unused
  __builtin_amdgcn_tensor_load_to_lds(g0, g1, z4, z4, z8, 0);
}

// ---- One PANEL of WMMA + exp folding, branch-free straight-line code ------
// MASK=true only for the single panel that contains the diagonal.
template <bool MASK>
__device__ __forceinline__ void panel_body(const _Float16* __restrict__ lbase,
                                           const v16h (&afrag)[8],
                                           const int (&rows)[8],
                                           float (&srun)[8],
                                           int cpan, int m, int hi) {
#pragma unroll
  for (int t = 0; t < PANEL / 16; ++t) {
    // B = E^T: column (c+m) of B is row (c+m) of E (now in LDS).
    // B-frag lane m holds col N=m, K = k0 + 16*hi + 0..15: one 32 B chunk.
    const _Float16* bp = lbase + (t * 16 + m) * DIM + hi * 16;
    v16h bfr[8];
#pragma unroll
    for (int k = 0; k < 8; ++k)                    // all ds_loads up front
      bfr[k] = *(const v16h*)(bp + k * 32);

    v8f acc = {};
#pragma unroll
    for (int k = 0; k < 8; ++k)
      acc = __builtin_amdgcn_wmma_f32_16x16x32_f16(
          /*neg_a=*/false, afrag[k], /*neg_b=*/false, bfr[k],
          /*c_mod=*/(short)0, acc, /*reuse_a=*/false, /*reuse_b=*/false);

    // Fold into running sums: exp(scale*(cos-1)).
    const int col = cpan + t * 16 + m;
#pragma unroll
    for (int r = 0; r < 8; ++r) {
      float ev = __expf(SCALE * (acc[r] - 1.0f));
      if (MASK) srun[r] += (rows[r] != col) ? ev : 0.0f;
      else      srun[r] += ev;
    }
  }
}

// -------- kernel 2: fused WMMA GEMM + fixed-max exp accumulation -----------
// Grid: (N_ROWS/64) * CSPLIT blocks of 128 threads (4 waves).
// Wave w handles 16 rows; the block covers a 1024-column chunk, streamed
// through LDS in PANEL-column double-buffered TDM panels.
__global__ void __launch_bounds__(128)
vit_simclr_lse_kernel(const _Float16* __restrict__ e16,
                      float* __restrict__ partial) {
  __shared__ _Float16 ldsp[2][PANEL * DIM];          // 2 x 32 KB

  const int lane  = threadIdx.x & 31;
  const int wave  = threadIdx.x >> 5;
  const int rb    = (blockIdx.x / CSPLIT) * 64;
  const int chunk =  blockIdx.x % CSPLIT;
  const int cbeg  = chunk * (N_ROWS / CSPLIT);

  const int m  = lane & 15;    // A-frag: row in tile; B-frag: col in tile
  const int hi = lane >> 4;    // half-wave selector (K sub-block)
  const int rowbase = rb + wave * 16;

  // Preload all A fragments for this wave's 16 rows (K = 0..255, 8 frags).
  // K order used here (same as B frags): lane m holds row M=m; elems 0-7 -> K=k0+hi*16+0..7,
  // elems 8-15 -> K=k0+hi*16+8+0..7: two contiguous 16 B chunks.
  v16h afrag[8];
  {
    const _Float16* ar = e16 + (size_t)(rowbase + m) * DIM + hi * 16;
#pragma unroll
    for (int k = 0; k < 8; ++k) {
      v8h lo = *(const v8h*)(ar + k * 32);
      v8h hh = *(const v8h*)(ar + k * 32 + 8);
      v16h a;
#pragma unroll
      for (int i = 0; i < 8; ++i) { a[i] = lo[i]; a[8 + i] = hh[i]; }
      afrag[k] = a;
    }
  }

  // C/D layout: VGPR r holds row (rowbase + r + 8*hi), column (c + m).
  int   rows[8];
  float srun[8];
#pragma unroll
  for (int r = 0; r < 8; ++r) { rows[r] = rowbase + r + 8 * hi; srun[r] = 0.0f; }

  const unsigned lds_off[2] = {
    (unsigned)(uintptr_t)(&ldsp[0][0]),
    (unsigned)(uintptr_t)(&ldsp[1][0]) };

  const int NPANEL = (N_ROWS / CSPLIT) / PANEL;      // 16 panels per chunk

  // Prologue: DMA panel 0.
  if (wave == 0) {
    tdm_load_panel(e16 + (size_t)cbeg * DIM, lds_off[0]);
    __builtin_amdgcn_s_wait_tensorcnt(0);
  }
  __syncthreads();

  for (int p = 0; p < NPANEL; ++p) {
    const int cur = p & 1;
    // Kick off DMA of next panel into the other buffer (overlaps compute).
    if (wave == 0 && p + 1 < NPANEL)
      tdm_load_panel(e16 + (size_t)(cbeg + (p + 1) * PANEL) * DIM,
                     lds_off[cur ^ 1]);

    const int cpan = cbeg + p * PANEL;
    // rb and cpan are both 64-aligned and rows span exactly [rb, rb+64):
    // the diagonal intersects exactly the panel with cpan == rb.
    if (cpan == rb)
      panel_body<true >(&ldsp[cur][0], afrag, rows, srun, cpan, m, hi);
    else
      panel_body<false>(&ldsp[cur][0], afrag, rows, srun, cpan, m, hi);

    // Publish the freshly DMA'd panel to all waves.
    if (wave == 0) __builtin_amdgcn_s_wait_tensorcnt(0);
    __syncthreads();
  }

  // Combine the 16 lanes of each half-wave (they cover the 16 column phases).
#pragma unroll
  for (int off = 1; off < 16; off <<= 1)
#pragma unroll
    for (int r = 0; r < 8; ++r)
      srun[r] += __shfl_xor(srun[r], off, 32);

  if (m == 0) {
#pragma unroll
    for (int r = 0; r < 8; ++r)
      partial[(size_t)rows[r] * CSPLIT + chunk] = srun[r];
  }
}

// ------- kernel 3: merge chunks, add log, subtract positive logit ----------
__global__ void vit_merge_kernel(const _Float16* __restrict__ e16,
                                 const float* __restrict__ partial,
                                 float* __restrict__ rowval) {
  const int i = blockIdx.x * blockDim.x + threadIdx.x;
  if (i >= N_ROWS) return;
  float S = 0.0f;
#pragma unroll
  for (int c = 0; c < CSPLIT; ++c) S += partial[(size_t)i * CSPLIT + c];
  const float lse = SCALE + logf(S);    // fixed max = 1/T added back

  const int j = (i + HALF_B) & (N_ROWS - 1);
  const v8h* a = (const v8h*)(e16 + (size_t)i * DIM);
  const v8h* b = (const v8h*)(e16 + (size_t)j * DIM);
  float dot = 0.0f;
  for (int k = 0; k < DIM / 8; ++k) {
    v8h x = a[k], y = b[k];
#pragma unroll
    for (int u = 0; u < 8; ++u) dot += (float)x[u] * (float)y[u];
  }
  rowval[i] = lse - SCALE * dot;
}

// -------- kernel 4: deterministic fixed-order mean reduction ---------------
__global__ void vit_reduce_kernel(const float* __restrict__ rowval,
                                  float* __restrict__ out) {
  __shared__ float red[256];
  const int t = threadIdx.x;
  float s = 0.0f;
  for (int i = t; i < N_ROWS; i += 256) s += rowval[i];   // fixed order
  red[t] = s;
  __syncthreads();
  for (int k = 128; k > 0; k >>= 1) {
    if (t < k) red[t] += red[t + k];
    __syncthreads();
  }
  if (t == 0) out[0] = red[0] / (float)N_ROWS;
}

// ---------------------------------------------------------------------------
extern "C" void kernel_launch(void* const* d_in, const int* in_sizes, int n_in,
                              void* d_out, int out_size, void* d_ws, size_t ws_size,
                              hipStream_t stream) {
  const float* emb = (const float*)d_in[0];
  float* out = (float*)d_out;

  char* ws = (char*)d_ws;
  _Float16* e16    = (_Float16*)ws;                                    // 4 MB
  float*    part   = (float*)(ws + (size_t)N_ROWS * DIM * 2);          // 256 KB
  float*    rowval = (float*)(ws + (size_t)N_ROWS * DIM * 2
                                 + (size_t)N_ROWS * CSPLIT * 4);       // 32 KB

  vit_normalize_kernel<<<N_ROWS, 256, 0, stream>>>(emb, e16);
  vit_simclr_lse_kernel<<<(N_ROWS / 64) * CSPLIT, 128, 0, stream>>>(e16, part);
  vit_merge_kernel<<<N_ROWS / 256, 256, 0, stream>>>(e16, part, rowval);
  vit_reduce_kernel<<<1, 256, 0, stream>>>(rowval, out);
}
